// PhysicsBiasedAttention_360777253402
// MI455X (gfx1250) — matrix operand order repaired and checked by reading, not compiled, not run on hardware
//
#include <hip/hip_runtime.h>
#include <hip/hip_bf16.h>
#include <math.h>
#include <stdint.h>

// ---------------- problem constants ----------------
#define BATCH 2
#define SEQ   2048
#define DMODEL 512
#define NHEAD 8
#define HDIM  64
#define NRES  16
#define LAM_C   1.0f
#define SCALE_C 0.125f   // 1/sqrt(64)

// ---------------- WMMA vector types ----------------
typedef __attribute__((ext_vector_type(16))) __bf16 v16bf;
typedef __attribute__((ext_vector_type(8)))  __bf16 v8bf;
typedef __attribute__((ext_vector_type(8)))  float  v8f;

union ABf { v16bf v; v8bf h[2]; __bf16 e[16]; };

__device__ __forceinline__ v8f v8f_zero() {
    v8f z = {0.f,0.f,0.f,0.f,0.f,0.f,0.f,0.f};
    return z;
}

// Wait for all of this wave's async (global<->LDS) transfers to land.
__device__ __forceinline__ void wait_async0() {
#if __has_builtin(__builtin_amdgcn_s_wait_asynccnt)
    __builtin_amdgcn_s_wait_asynccnt(0);
#else
    asm volatile("s_wait_asynccnt 0x0" ::: "memory");
#endif
}

// Per-lane async 16B copy: global -> LDS (tracked on ASYNCcnt).
__device__ __forceinline__ void async_copy16(const void* gsrc, void* lds_dst) {
    unsigned lds = (unsigned)(uintptr_t)lds_dst;  // low 32 bits = LDS offset
    uint64_t ga  = (uint64_t)(uintptr_t)gsrc;
    asm volatile("global_load_async_to_lds_b128 %0, %1, off"
                 :: "v"(lds), "v"(ga) : "memory");
}

// ---------------- f32 -> bf16 convert ----------------
__global__ void cvt_bf16_kernel(const float* __restrict__ src,
                                __bf16* __restrict__ dst, int n) {
    int i = blockIdx.x * blockDim.x + threadIdx.x;
    if (i < n) dst[i] = (__bf16)src[i];
}

// ---------------- rb = ||pde_residuals||_2 per token ----------------
__global__ void rb_kernel(const float* __restrict__ res,
                          float* __restrict__ rb, int n) {
    int i = blockIdx.x * blockDim.x + threadIdx.x;
    if (i < n) {
        float s = 0.f;
        #pragma unroll
        for (int j = 0; j < NRES; ++j) {
            float v = res[(size_t)i * NRES + j];
            s += v * v;
        }
        rb[i] = sqrtf(s);
    }
}

// ---------------- GEMM: Y[M,N] = A[M,K](bf16) @ W[N,K]^T(bf16) + bias ----
// One wave computes a 16(M) x 64(N) strip. grid = (N/64, M/128), block = 256.
template <bool F32OUT>
__global__ __launch_bounds__(256) void gemm_wmma_kernel(
    const __bf16* __restrict__ A, const __bf16* __restrict__ W,
    const float* __restrict__ bias, void* __restrict__ Y,
    int M, int Ncols, int K) {
    const int wave = threadIdx.x >> 5;
    const int lane = threadIdx.x & 31;
    const int lo = lane & 15;
    const int hi = lane >> 4;

    const int m0 = blockIdx.y * 128 + wave * 16;
    const int n0 = blockIdx.x * 64;

    v8f acc[4];
    #pragma unroll
    for (int t = 0; t < 4; ++t) acc[t] = v8f_zero();

    // A fragment: lane lo -> row m0+lo; chunks at k + hi*16 and k + hi*16 + 8
    const __bf16* arow = A + (size_t)(m0 + lo) * K + hi * 16;

    for (int ks = 0; ks < K; ks += 32) {
        ABf a;
        a.h[0] = *(const v8bf*)(arow + ks);
        a.h[1] = *(const v8bf*)(arow + ks + 8);
        #pragma unroll
        for (int nt = 0; nt < 4; ++nt) {
            // B fragment: lane lo -> column n0+nt*16+lo, K rows ks+hi*16 .. +16
            const __bf16* wrow = W + (size_t)(n0 + nt * 16 + lo) * K + ks + hi * 16;
            v16bf bfrag = *(const v16bf*)wrow;
            acc[nt] = __builtin_amdgcn_wmma_f32_16x16x32_bf16(
                false, a.v, false, bfrag, (short)0, acc[nt], false, false);
        }
    }

    // bias add + store.  C layout: VGPR r, lane c -> row m0 + r + 8*hi, col n0+nt*16+lo
    #pragma unroll
    for (int nt = 0; nt < 4; ++nt) {
        const int col = n0 + nt * 16 + lo;
        const float bv = bias[col];
        #pragma unroll
        for (int r = 0; r < 8; ++r) {
            const int row = m0 + r + 8 * hi;
            const float val = acc[nt][r] + bv;
            if (F32OUT)
                ((float*)Y)[(size_t)row * Ncols + col] = val;
            else
                ((__bf16*)Y)[(size_t)row * Ncols + col] = (__bf16)val;
        }
    }
}

// ---------------- Flash attention core ----------------
// grid = (SEQ/128, NHEAD, BATCH), block = 256 (8 waves).  Wave owns 16 queries.
// S^T = K_tile @ Q^T per 16x16 tile so the online softmax is per-lane.
// Double-buffered LDS staging: K tile via async global->LDS, V tile transposed
// through VGPRs, one barrier per key step.
__global__ __launch_bounds__(256) void flash_attn_kernel(
    const __bf16* __restrict__ qb, const __bf16* __restrict__ kb,
    const __bf16* __restrict__ vb, const float* __restrict__ rb,
    const int* __restrict__ mask, __bf16* __restrict__ ob) {
    const int b = blockIdx.z;
    const int h = blockIdx.y;
    const int wave = threadIdx.x >> 5;
    const int lane = threadIdx.x & 31;
    const int lo = lane & 15;
    const int hi = lane >> 4;
    const int q0 = blockIdx.x * 128 + wave * 16;

    __shared__ __align__(32) __bf16 lsK[2][32 * HDIM];   // K tiles, row-major [32][64]
    __shared__ __align__(32) __bf16 lsVt[2][HDIM * 32];  // V tiles transposed [64][32]
    __shared__ float lsRb[2][32];

    const int t = threadIdx.x;
    const int srow = t >> 3;  // staging: key row 0..31
    const int sseg = t & 7;   // staging: 8-bf16 segment

    // Q B-fragments (column q = Q row q0+q). Two d-steps (d 0..31 / 32..63).
    v16bf qf[2];
    {
        const __bf16* qrow = qb + ((size_t)(b * SEQ + q0 + lo)) * DMODEL + h * HDIM + hi * 16;
        qf[0] = *(const v16bf*)(qrow);
        qf[1] = *(const v16bf*)(qrow + 32);
    }

    const float rbq_term = LAM_C * rb[b * SEQ + q0 + lo]; // + lam*rb[q]
    const int* mrow = mask + (size_t)b * SEQ * SEQ + (size_t)(q0 + lo) * SEQ;

    float m_run = -1e30f;
    float l_run = 0.f;
    v8f o[4];
    #pragma unroll
    for (int i = 0; i < 4; ++i) o[i] = v8f_zero();

    auto stage = [&](int kb0, int buf) {
        const size_t gtok = (size_t)(b * SEQ + kb0 + srow) * DMODEL + h * HDIM + sseg * 8;
        // K tile: 16B per lane, async straight into LDS (row-major)
        async_copy16(kb + gtok, &lsK[buf][srow * HDIM + sseg * 8]);
        // V tile: transpose into lsVt[buf][d][k] through VGPRs
        v8bf vv = *(const v8bf*)(vb + gtok);
        #pragma unroll
        for (int j = 0; j < 8; ++j)
            lsVt[buf][(sseg * 8 + j) * 32 + srow] = vv[j];
        if (t < 32) lsRb[buf][t] = rb[b * SEQ + kb0 + t];
    };

    const int NSTEP = SEQ / 32;
    stage(0, 0);
    wait_async0();
    __syncthreads();

    for (int step = 0; step < NSTEP; ++step) {
        const int cur = step & 1;
        const int kb0 = step * 32;

        // Prefetch next tile into the other buffer while we compute.
        if (step + 1 < NSTEP) stage(kb0 + 32, cur ^ 1);

        // S^T tiles: keys (kb0 + tt*16 .. +16) x 16 queries, accum over d
        v8f st[2];
        st[0] = v8f_zero();
        st[1] = v8f_zero();
        #pragma unroll
        for (int tt = 0; tt < 2; ++tt) {
            #pragma unroll
            for (int s = 0; s < 2; ++s) {
                ABf a; // A = K rows (16 x 32 over d)
                const __bf16* kp = &lsK[cur][(tt * 16 + lo) * HDIM + s * 32 + hi * 16];
                a.h[0] = *(const v8bf*)kp;
                a.h[1] = *(const v8bf*)(kp + 8);
                st[tt] = __builtin_amdgcn_wmma_f32_16x16x32_bf16(
                    false, a.v, false, qf[s], (short)0, st[tt], false, false);
            }
        }

        // Per-lane: scores for query (q0+lo), keys kb0 + tt*16 + 8*hi + r
        float ptile[16];
        float mloc = -1e30f;
        #pragma unroll
        for (int tt = 0; tt < 2; ++tt) {
            #pragma unroll
            for (int r = 0; r < 8; ++r) {
                const int koff = tt * 16 + 8 * hi + r;
                float s = st[tt][r] * SCALE_C - LAM_C * lsRb[cur][koff] + rbq_term;
                s = (mrow[kb0 + koff] == 0) ? -1e30f : s;
                ptile[tt * 8 + r] = s;
                mloc = fmaxf(mloc, s);
            }
        }
        mloc = fmaxf(mloc, __shfl_xor(mloc, 16, 32)); // merge the two key-halves
        const float m_new = fmaxf(m_run, mloc);
        const float alpha = __expf(m_run - m_new);
        m_run = m_new;

        float lsum = 0.f;
        #pragma unroll
        for (int i = 0; i < 16; ++i) {
            float p = __expf(ptile[i] - m_new);
            ptile[i] = p;
            lsum += p;
        }
        l_run = l_run * alpha + lsum;

        // P^T C-layout == bf16 A-fragment layout: pure in-lane repack.
        ABf pa;
        #pragma unroll
        for (int r = 0; r < 8; ++r) {
            pa.e[r]     = (__bf16)ptile[r];       // keys tt=0 half -> K {0..7}/{8..15}
            pa.e[8 + r] = (__bf16)ptile[8 + r];   // keys tt=1 half -> K {16..23}/{24..31}
        }

        // Rescale O rows by alpha (row r+8*hi lives at lane r / r+8 in softmax layout)
        #pragma unroll
        for (int r = 0; r < 8; ++r) {
            const float ar = __shfl(alpha, hi ? (8 + r) : r, 32);
            o[0][r] *= ar; o[1][r] *= ar; o[2][r] *= ar; o[3][r] *= ar;
        }

        // O += P @ V   (B fragment column d from transposed V tile)
        #pragma unroll
        for (int dt = 0; dt < 4; ++dt) {
            const __bf16* vp = &lsVt[cur][(dt * 16 + lo) * 32 + hi * 8];
            ABf vf;
            vf.h[0] = *(const v8bf*)vp;
            vf.h[1] = *(const v8bf*)(vp + 16);
            o[dt] = __builtin_amdgcn_wmma_f32_16x16x32_bf16(
                false, pa.v, false, vf.v, (short)0, o[dt], false, false);
        }

        // Next buffer must be fully staged (our async K + everyone's V stores)
        // before the next iteration reads it; our LDS reads of `cur` are done
        // before we signal (dscnt) so next iteration may overwrite it.
        wait_async0();
        __syncthreads();
    }

    // Finalize: divide by row sums, write bf16 O.
    const float l_tot = l_run + __shfl_xor(l_run, 16, 32);
    const float inv = 1.0f / l_tot;
    #pragma unroll
    for (int r = 0; r < 8; ++r) {
        const float ir = __shfl(inv, hi ? (8 + r) : r, 32);
        const int row = q0 + r + 8 * hi;
        __bf16* orow = ob + ((size_t)(b * SEQ + row)) * DMODEL + h * HDIM + lo;
        #pragma unroll
        for (int dt = 0; dt < 4; ++dt)
            orow[dt * 16] = (__bf16)(o[dt][r] * ir);
    }
}

// ---------------- host side ----------------
extern "C" void kernel_launch(void* const* d_in, const int* in_sizes, int n_in,
                              void* d_out, int out_size, void* d_ws, size_t ws_size,
                              hipStream_t stream) {
    (void)in_sizes; (void)n_in; (void)out_size; (void)ws_size;

    const float* x   = (const float*)d_in[0];
    const float* res = (const float*)d_in[1];
    const int*   msk = (const int*)d_in[2];
    const float* Wq  = (const float*)d_in[3];
    const float* bq  = (const float*)d_in[4];
    const float* Wk  = (const float*)d_in[5];
    const float* bk  = (const float*)d_in[6];
    const float* Wv  = (const float*)d_in[7];
    const float* bv  = (const float*)d_in[8];
    const float* Wo  = (const float*)d_in[9];
    const float* bo  = (const float*)d_in[10];

    const int M = BATCH * SEQ;           // 4096
    const int TOK = M * DMODEL;          // 2,097,152
    const int WEL = DMODEL * DMODEL;     // 262,144

    char* ws = (char*)d_ws;
    size_t off = 0;
    __bf16* xb  = (__bf16*)(ws + off); off += (size_t)TOK * 2;
    __bf16* wqb = (__bf16*)(ws + off); off += (size_t)WEL * 2;
    __bf16* wkb = (__bf16*)(ws + off); off += (size_t)WEL * 2;
    __bf16* wvb = (__bf16*)(ws + off); off += (size_t)WEL * 2;
    __bf16* wob = (__bf16*)(ws + off); off += (size_t)WEL * 2;
    __bf16* qb  = (__bf16*)(ws + off); off += (size_t)TOK * 2;
    __bf16* kbf = (__bf16*)(ws + off); off += (size_t)TOK * 2;
    __bf16* vbf = (__bf16*)(ws + off); off += (size_t)TOK * 2;
    __bf16* ob  = (__bf16*)(ws + off); off += (size_t)TOK * 2;
    float*  rbw = (float*)(ws + off);  off += (size_t)M * 4;

    // 1) converts + rb
    cvt_bf16_kernel<<<(TOK + 255) / 256, 256, 0, stream>>>(x, xb, TOK);
    cvt_bf16_kernel<<<(WEL + 255) / 256, 256, 0, stream>>>(Wq, wqb, WEL);
    cvt_bf16_kernel<<<(WEL + 255) / 256, 256, 0, stream>>>(Wk, wkb, WEL);
    cvt_bf16_kernel<<<(WEL + 255) / 256, 256, 0, stream>>>(Wv, wvb, WEL);
    cvt_bf16_kernel<<<(WEL + 255) / 256, 256, 0, stream>>>(Wo, wob, WEL);
    rb_kernel<<<(M + 255) / 256, 256, 0, stream>>>(res, rbw, M);

    // 2) QKV projections
    dim3 ggrid(DMODEL / 64, M / 128);
    gemm_wmma_kernel<false><<<ggrid, 256, 0, stream>>>(xb, wqb, bq, qb,  M, DMODEL, DMODEL);
    gemm_wmma_kernel<false><<<ggrid, 256, 0, stream>>>(xb, wkb, bk, kbf, M, DMODEL, DMODEL);
    gemm_wmma_kernel<false><<<ggrid, 256, 0, stream>>>(xb, wvb, bv, vbf, M, DMODEL, DMODEL);

    // 3) attention
    dim3 agrid(SEQ / 128, NHEAD, BATCH);
    flash_attn_kernel<<<agrid, 256, 0, stream>>>(qb, kbf, vbf, rbw, msk, ob);

    // 4) output projection -> f32 d_out
    gemm_wmma_kernel<true><<<ggrid, 256, 0, stream>>>(ob, wob, bo, (float*)d_out, M, DMODEL, DMODEL);
}
